// LSTMModel_63943473102903
// MI455X (gfx1250) — matrix operand order repaired and checked by reading, not compiled, not run on hardware
//
#include <hip/hip_runtime.h>
#include <hip/hip_bf16.h>

// Problem constants (from reference): B=2048, T=512, H=64, gates G=4H=256
#define BATCH 2048
#define TLEN  512
#define H     64
#define G     256
#define BT    16            // batch rows per workgroup (WMMA M)
#define NWG   (BATCH / BT)  // 128 workgroups

typedef __bf16 bf16_t;
typedef __attribute__((ext_vector_type(16))) __bf16 v16bf;
typedef __attribute__((ext_vector_type(8)))  __bf16 v8bf;
typedef __attribute__((ext_vector_type(8)))  float  v8f;
typedef __attribute__((ext_vector_type(4)))  float  v4f;

struct __align__(16) SharedMem {
  float  gates[G][BT];   // 16 KB : gate pre-activations, [N][M] (column-major vs math view)
  bf16_t h0[BT][H];      // 2 KB  : layer-0 hidden state (bf16, feeds WMMA A operand)
  bf16_t h1[BT][H];      // 2 KB  : layer-1 hidden state
  float  xcol[BT];       // 64 B  : x[:, t] for this batch tile
};

__device__ __forceinline__ float sigm_f(float v) {
  return 1.0f / (1.0f + __expf(-v));
}
__device__ __forceinline__ float tanh_f(float v) {
  float e = __expf(-2.0f * v);
  return (1.0f - e) / (1.0f + e);
}

// A operand: 16x32 bf16 tile from LDS h-buffer [BT][H].
// Lane l (l<16): row M=l, K = K0 + 0..15; lanes>=16: row M=l-16, K = K0 + 16..31.
__device__ __forceinline__ v16bf load_a_tile(const bf16_t* hb, int arow, int akoff, int K0) {
  const bf16_t* p = hb + arow * H + K0 + akoff;       // 16B-aligned
  v8bf lo = *(const v8bf*)(p);
  v8bf hi = *(const v8bf*)(p + 8);
  v16bf a;
#pragma unroll
  for (int e = 0; e < 8; ++e) { a[e] = lo[e]; a[8 + e] = hi[e]; }
  return a;
}

// C/D layout: VGPR r, lane l -> M = r + (l>=16 ? 8 : 0), N = n0 + (l&15).
// gates stored [N][M] so the 8 accumulator regs are 8 contiguous floats -> two b128 stores.
__device__ __forceinline__ void store_gates(float* gp, v8f acc) {
  v4f lo = { acc[0], acc[1], acc[2], acc[3] };
  v4f hi = { acc[4], acc[5], acc[6], acc[7] };
  *(v4f*)(gp)     = lo;
  *(v4f*)(gp + 4) = hi;
}

#define WMMA_BF16(A, Bm, C) \
  __builtin_amdgcn_wmma_f32_16x16x32_bf16(false, (A), false, (Bm), (short)0, (C), false, false)

__global__ __launch_bounds__(256, 1)
void lstm2_fused_kernel(const float* __restrict__ x,
                        const float* __restrict__ w_ih0,
                        const float* __restrict__ w_hh0,
                        const float* __restrict__ b_ih0,
                        const float* __restrict__ b_hh0,
                        const float* __restrict__ w_ih1,
                        const float* __restrict__ w_hh1,
                        const float* __restrict__ b_ih1,
                        const float* __restrict__ b_hh1,
                        const float* __restrict__ fc_w,
                        const float* __restrict__ fc_b,
                        float* __restrict__ out)
{
  __shared__ SharedMem sm;
  const int tid   = threadIdx.x;
  const int lane  = tid & 31;
  const int wave  = tid >> 5;          // 8 waves; wave owns N columns [wave*32, wave*32+32)
  const int bbase = blockIdx.x * BT;

  // ---- B-operand lane mapping (K x N = 32 x 16 bf16) ----
  // lane<16: N=lane,   K = K0 + 0..15 ; lane>=16: N=lane-16, K = K0 + 16..31
  const int bn = lane & 15;
  const int bk = (lane >> 4) * 16;

  // Loop-invariant weights -> bf16 fragments held in VGPRs for all 512 steps.
  // [matrix: whh0, wih1, whh1][N-tile j][K-chunk c]  (weights are row-major [N=256][K=64])
  v16bf Bw[3][2][2];
  {
    const float* Wmat[3] = { w_hh0, w_ih1, w_hh1 };
#pragma unroll
    for (int m = 0; m < 3; ++m) {
#pragma unroll
      for (int j = 0; j < 2; ++j) {
        const int n = (2 * wave + j) * 16 + bn;
#pragma unroll
        for (int c = 0; c < 2; ++c) {
          const float* src = Wmat[m] + n * H + c * 32 + bk;  // 16 contiguous f32 (64B)
          v16bf b;
#pragma unroll
          for (int e = 0; e < 16; ++e) b[e] = (bf16_t)src[e];
          Bw[m][j][c] = b;
        }
      }
    }
  }

  // ---- Elementwise-phase constants: thread owns hidden unit en, rows erb..erb+3 ----
  const int en  = tid & 63;
  const int erb = (tid >> 6) * 4;
  float bsum0[4], bsum1[4], wi0[4];
#pragma unroll
  for (int g = 0; g < 4; ++g) {        // gate order i,f,g,o at N = en + 64*g
    bsum0[g] = b_ih0[en + 64 * g] + b_hh0[en + 64 * g];
    bsum1[g] = b_ih1[en + 64 * g] + b_hh1[en + 64 * g];
    wi0[g]   = w_ih0[en + 64 * g];     // w_ih0 is (256,1)
  }
  float c0[4] = {0.f, 0.f, 0.f, 0.f};
  float c1[4] = {0.f, 0.f, 0.f, 0.f};

  // zero h state
  for (int i = tid; i < BT * H; i += 256) {
    (&sm.h0[0][0])[i] = (bf16_t)0.0f;
    (&sm.h1[0][0])[i] = (bf16_t)0.0f;
  }
  __syncthreads();

  const int arow  = lane & 15;
  const int akoff = (lane >> 4) * 16;
  const int gmb   = (lane >> 4) * 8;   // M base for C/D store

  for (int t = 0; t < TLEN; ++t) {
    // stage x[:, t] for this batch tile (layer-0 input term is rank-1)
    if (tid < BT) {
      sm.xcol[tid] = x[(size_t)(bbase + tid) * TLEN + t];
      if (t + 16 < TLEN)
        __builtin_prefetch(&x[(size_t)(bbase + tid) * TLEN + t + 16], 0, 0);
    }

    // ================= layer 0: gates = h0 @ w_hh0^T =================
    {
      v16bf a0 = load_a_tile(&sm.h0[0][0], arow, akoff, 0);
      v16bf a1 = load_a_tile(&sm.h0[0][0], arow, akoff, 32);
#pragma unroll
      for (int j = 0; j < 2; ++j) {
        v8f acc = {};
        acc = WMMA_BF16(a0, Bw[0][j][0], acc);
        acc = WMMA_BF16(a1, Bw[0][j][1], acc);
        store_gates(&sm.gates[(2 * wave + j) * 16 + bn][gmb], acc);
      }
    }
    __syncthreads();

    // layer-0 LSTM cell (adds x*w_ih0 + biases here)
    {
      v4f gi = *(const v4f*)&sm.gates[en      ][erb];
      v4f gf = *(const v4f*)&sm.gates[en +  64][erb];
      v4f gg = *(const v4f*)&sm.gates[en + 128][erb];
      v4f go = *(const v4f*)&sm.gates[en + 192][erb];
#pragma unroll
      for (int k = 0; k < 4; ++k) {
        float xv = sm.xcol[erb + k];
        float iv = sigm_f(gi[k] + xv * wi0[0] + bsum0[0]);
        float fv = sigm_f(gf[k] + xv * wi0[1] + bsum0[1]);
        float gv = tanh_f(gg[k] + xv * wi0[2] + bsum0[2]);
        float ov = sigm_f(go[k] + xv * wi0[3] + bsum0[3]);
        c0[k] = fv * c0[k] + iv * gv;
        sm.h0[erb + k][en] = (bf16_t)(ov * tanh_f(c0[k]));
      }
    }
    __syncthreads();

    // ============ layer 1: gates = h0 @ w_ih1^T + h1 @ w_hh1^T ============
    {
      v16bf ai0 = load_a_tile(&sm.h0[0][0], arow, akoff, 0);
      v16bf ai1 = load_a_tile(&sm.h0[0][0], arow, akoff, 32);
      v16bf ah0 = load_a_tile(&sm.h1[0][0], arow, akoff, 0);
      v16bf ah1 = load_a_tile(&sm.h1[0][0], arow, akoff, 32);
#pragma unroll
      for (int j = 0; j < 2; ++j) {
        v8f acc = {};
        acc = WMMA_BF16(ai0, Bw[1][j][0], acc);
        acc = WMMA_BF16(ai1, Bw[1][j][1], acc);
        acc = WMMA_BF16(ah0, Bw[2][j][0], acc);
        acc = WMMA_BF16(ah1, Bw[2][j][1], acc);
        store_gates(&sm.gates[(2 * wave + j) * 16 + bn][gmb], acc);
      }
    }
    __syncthreads();

    // layer-1 LSTM cell
    {
      v4f gi = *(const v4f*)&sm.gates[en      ][erb];
      v4f gf = *(const v4f*)&sm.gates[en +  64][erb];
      v4f gg = *(const v4f*)&sm.gates[en + 128][erb];
      v4f go = *(const v4f*)&sm.gates[en + 192][erb];
#pragma unroll
      for (int k = 0; k < 4; ++k) {
        float iv = sigm_f(gi[k] + bsum1[0]);
        float fv = sigm_f(gf[k] + bsum1[1]);
        float gv = tanh_f(gg[k] + bsum1[2]);
        float ov = sigm_f(go[k] + bsum1[3]);
        c1[k] = fv * c1[k] + iv * gv;
        sm.h1[erb + k][en] = (bf16_t)(ov * tanh_f(c1[k]));
      }
    }
    __syncthreads();
  }

  // ---- final FC on last-timestep layer-1 hidden state: out[b] = h1[b,:] . fc_w + fc_b ----
  if (tid < BT) {
    float acc = fc_b[0];
#pragma unroll
    for (int n = 0; n < H; ++n)
      acc += (float)sm.h1[tid][n] * fc_w[n];
    out[bbase + tid] = acc;
  }
}

extern "C" void kernel_launch(void* const* d_in, const int* in_sizes, int n_in,
                              void* d_out, int out_size, void* d_ws, size_t ws_size,
                              hipStream_t stream) {
  (void)in_sizes; (void)n_in; (void)out_size; (void)d_ws; (void)ws_size;
  const float* xin   = (const float*)d_in[0];
  const float* wih0  = (const float*)d_in[1];
  const float* whh0  = (const float*)d_in[2];
  const float* bih0  = (const float*)d_in[3];
  const float* bhh0  = (const float*)d_in[4];
  const float* wih1  = (const float*)d_in[5];
  const float* whh1  = (const float*)d_in[6];
  const float* bih1  = (const float*)d_in[7];
  const float* bhh1  = (const float*)d_in[8];
  const float* fcw   = (const float*)d_in[9];
  const float* fcb   = (const float*)d_in[10];
  float* outp        = (float*)d_out;

  lstm2_fused_kernel<<<dim3(NWG), dim3(256), 0, stream>>>(
      xin, wih0, whh0, bih0, bhh0, wih1, whh1, bih1, bhh1, fcw, fcb, outp);
}
